// EvoformerBlock_43044162241217
// MI455X (gfx1250) — matrix operand order repaired and checked by reading, not compiled, not run on hardware
//
#include <hip/hip_runtime.h>
#include <hip/hip_bf16.h>

typedef __attribute__((ext_vector_type(16))) __bf16 v16bf;
typedef __attribute__((ext_vector_type(8)))  float  v8f;

// ---------------------------------------------------------------------------
// Generic strided + 2-level-batched GEMM, bf16 WMMA, f32 accumulate.
// Block = 128 threads (4 wave32). Macro tile 64(M) x 32(N); each wave owns a
// 16x32 strip (two f32 16x16 accumulators). K staged in 64-wide LDS tiles,
// two v_wmma_f32_16x16x32_bf16 chunks per strip per stage => 4 WMMA/wave/stage.
// Y[b1,b0][m,n] = act(alpha * sum_k X[m,k]*W[k,n] + bias[n]) (or += if accum)
// X element addr: xoff + m*xRM + k*xRK ; W: woff + k*wRK + n*wRN
// batch: b0 = blockIdx.z % B0, b1 = blockIdx.z / B0
// ---------------------------------------------------------------------------
__global__ void gemm_kernel(const float* X, const float* W, const float* bias,
                            float* Y, int M, int Nc, int K,
                            long xRM, long xRK, long wRK, long wRN,
                            long yRM, long yRN,
                            long xB0, long xB1, long wB0, long wB1,
                            long yB0, long yB1, int B0,
                            float alpha, int act, int accum)
{
    __shared__ float sX[64 * 64]; // [m][k]
    __shared__ float sW[64 * 32]; // [k][n]
    int tid  = threadIdx.x;
    int lane = tid & 31;
    int wid  = tid >> 5;
    int bz   = blockIdx.z;
    int b0   = bz % B0;
    int b1   = bz / B0;
    long xoff = (long)b0 * xB0 + (long)b1 * xB1;
    long woff = (long)b0 * wB0 + (long)b1 * wB1;
    long yoff = (long)b0 * yB0 + (long)b1 * yB1;

    int mTile = blockIdx.x * 64;
    int nTile = blockIdx.y * 32;
    int subM  = wid * 16;          // wave's row strip within macro tile
    int half  = lane >> 4;
    int mr    = lane & 15;

    v8f acc0 = {0.f,0.f,0.f,0.f,0.f,0.f,0.f,0.f};
    v8f acc1 = {0.f,0.f,0.f,0.f,0.f,0.f,0.f,0.f};

    for (int k0 = 0; k0 < K; k0 += 64) {
        // ---- stage X tile 64x64 (branch-free: clamp + select) ----
        for (int t = tid; t < 64 * 64; t += 128) {
            int r = t >> 6, c = t & 63;
            int gm = mTile + r, gk = k0 + c;
            int gmc = gm < M ? gm : M - 1;
            int gkc = gk < K ? gk : K - 1;
            float xv = X[xoff + (long)gmc * xRM + (long)gkc * xRK];
            sX[t] = (gm < M && gk < K) ? xv : 0.f;
        }
        // ---- stage W tile 64x32 ----
        for (int t = tid; t < 64 * 32; t += 128) {
            int r = t >> 5, c = t & 31;
            int gk = k0 + r, gn = nTile + c;
            int gkc = gk < K ? gk : K - 1;
            int gnc = gn < Nc ? gn : Nc - 1;
            float wv = W[woff + (long)gkc * wRK + (long)gnc * wRN];
            sW[t] = (gk < K && gn < Nc) ? wv : 0.f;
        }
        // prefetch next K tile of X into cache (global_prefetch_b8)
        if (k0 + 64 < K) {
            int pm = mTile + (tid & 63);
            if (pm < M)
                __builtin_prefetch(X + xoff + (long)pm * xRM + (long)(k0 + 64) * xRK, 0, 1);
        }
        __syncthreads();

        int am = subM + mr;        // row in sX for this wave
#pragma unroll
        for (int c0 = 0; c0 < 64; c0 += 32) {
            v16bf av, bv0, bv1;
#pragma unroll
            for (int v = 0; v < 8; ++v) {
                // A 16x32 bf16, this code's k order (same as B): lanes0-15 K=0..15 (2/VGPR), lanes16-31 K=16..31
                int kA = 2 * v + half * 16 + c0;
                av[2 * v]     = (__bf16)sX[am * 64 + kA];
                av[2 * v + 1] = (__bf16)sX[am * 64 + kA + 1];
                // B 32x16 bf16, this code's k order (same as A): lanes0-15 K=0..15 (2/VGPR), lanes16-31 K=16..31
                int kB = 2 * v + 16 * half + c0;
                bv0[2 * v]     = (__bf16)sW[kB * 32 + mr];
                bv0[2 * v + 1] = (__bf16)sW[(kB + 1) * 32 + mr];
                bv1[2 * v]     = (__bf16)sW[kB * 32 + 16 + mr];
                bv1[2 * v + 1] = (__bf16)sW[(kB + 1) * 32 + 16 + mr];
            }
            acc0 = __builtin_amdgcn_wmma_f32_16x16x32_bf16(false, av, false, bv0,
                                                           (short)0, acc0, false, false);
            acc1 = __builtin_amdgcn_wmma_f32_16x16x32_bf16(false, av, false, bv1,
                                                           (short)0, acc1, false, false);
        }
        __syncthreads();
    }

    // ---- epilogue ----
    int n0 = nTile + mr;
    int n1 = nTile + 16 + mr;
#pragma unroll
    for (int r = 0; r < 8; ++r) {
        int m = mTile + subM + r + 8 * half; // C/D layout: VGPR r -> M = r + 8*half
        if (m < M) {
            if (n0 < Nc) {
                float val = alpha * acc0[r];
                if (bias) val += bias[n0];
                if (act == 1) val = val > 0.f ? val : 0.f;
                else if (act == 2) val = 1.f / (1.f + __expf(-val));
                long addr = yoff + (long)m * yRM + (long)n0 * yRN;
                if (accum) Y[addr] += val; else Y[addr] = val;
            }
            if (n1 < Nc) {
                float val = alpha * acc1[r];
                if (bias) val += bias[n1];
                if (act == 1) val = val > 0.f ? val : 0.f;
                else if (act == 2) val = 1.f / (1.f + __expf(-val));
                long addr = yoff + (long)m * yRM + (long)n1 * yRN;
                if (accum) Y[addr] += val; else Y[addr] = val;
            }
        }
    }
}

// ---------------------------------------------------------------------------
// LayerNorm over last dim C (64 threads per row)
// ---------------------------------------------------------------------------
__global__ void ln_kernel(const float* x, const float* g, const float* b,
                          float* y, int C)
{
    long row = blockIdx.x;
    const float* xr = x + row * (long)C;
    __shared__ float red[64];
    int tid = threadIdx.x;
    float s = 0.f;
    for (int c = tid; c < C; c += 64) s += xr[c];
    red[tid] = s; __syncthreads();
    for (int o = 32; o > 0; o >>= 1) { if (tid < o) red[tid] += red[tid + o]; __syncthreads(); }
    float mu = red[0] / C; __syncthreads();
    float vv = 0.f;
    for (int c = tid; c < C; c += 64) { float d = xr[c] - mu; vv += d * d; }
    red[tid] = vv; __syncthreads();
    for (int o = 32; o > 0; o >>= 1) { if (tid < o) red[tid] += red[tid + o]; __syncthreads(); }
    float rstd = rsqrtf(red[0] / C + 1e-5f);
    __syncthreads();
    for (int c = tid; c < C; c += 64)
        y[row * (long)C + c] = (xr[c] - mu) * rstd * g[c] + b[c];
}

// ---------------------------------------------------------------------------
// Generic gated attention for one (r0, r1, h) row. Context length nCtx <= 192.
// qoff = r0*q0 + r1*q1 + h*32 ; k/v ctx j at r0*k0 + r1*k1 + j*kc + h*32
// bias idx = r0*bb0 + r1*bb1 + j*bbc + h ; mask idx = r0*m0 + r1*m1 + j*mc
// out written at qoff (may alias Q; each block's Q region is private).
// ---------------------------------------------------------------------------
__global__ void attn_kernel(const float* Q, const float* K, const float* V,
                            const float* G, const float* bias, const float* mask,
                            float* out, int nCtx,
                            long q0, long q1, long k0, long k1, long kc,
                            long bb0, long bb1, long bbc,
                            long m0, long m1, long mc, float scale)
{
    int r0 = blockIdx.x, r1 = blockIdx.y, h = blockIdx.z;
    long qoff  = (long)r0 * q0 + (long)r1 * q1 + (long)h * 32;
    long kbase = (long)r0 * k0 + (long)r1 * k1 + (long)h * 32;
    __shared__ float sm[192];
    __shared__ float qs[32];
    __shared__ float red[64];
    int tid = threadIdx.x; // 64 threads
    if (tid < 32) qs[tid] = Q[qoff + tid];
    __syncthreads();
    for (int j = tid; j < nCtx; j += 64) {
        const float* kp = K + kbase + (long)j * kc;
        float a = 0.f;
        for (int c = 0; c < 32; ++c) a += qs[c] * kp[c];
        a *= scale;
        if (bias) a += bias[(long)r0 * bb0 + (long)r1 * bb1 + (long)j * bbc + h];
        if (mask) a += 1e9f * (mask[(long)r0 * m0 + (long)r1 * m1 + (long)j * mc] - 1.f);
        sm[j] = a;
    }
    __syncthreads();
    float mx = -1e30f;
    for (int j = tid; j < nCtx; j += 64) mx = fmaxf(mx, sm[j]);
    red[tid] = mx; __syncthreads();
    for (int o = 32; o > 0; o >>= 1) { if (tid < o) red[tid] = fmaxf(red[tid], red[tid + o]); __syncthreads(); }
    mx = red[0]; __syncthreads();
    float sum = 0.f;
    for (int j = tid; j < nCtx; j += 64) { float e = __expf(sm[j] - mx); sm[j] = e; sum += e; }
    red[tid] = sum; __syncthreads();
    for (int o = 32; o > 0; o >>= 1) { if (tid < o) red[tid] += red[tid + o]; __syncthreads(); }
    float inv = 1.f / red[0];
    __syncthreads();
    if (tid < 32) {
        float a = 0.f;
        for (int j = 0; j < nCtx; ++j) a += sm[j] * V[kbase + (long)j * kc + tid];
        a *= inv;
        float gt = G[qoff + tid];
        gt = 1.f / (1.f + __expf(-gt));
        out[qoff + tid] = gt * a;
    }
}

// p[i] = mask[i / C] * gsig[i] * p[i]
__global__ void gate_mask_mul(float* p, const float* gsig, const float* mask,
                              int C, long total)
{
    long i = (long)blockIdx.x * blockDim.x + threadIdx.x;
    if (i < total) p[i] = mask[i / C] * gsig[i] * p[i];
}

// z[i] += gsig[i] * t[i]
__global__ void gated_add(float* z, const float* gsig, const float* t, long total)
{
    long i = (long)blockIdx.x * blockDim.x + threadIdx.x;
    if (i < total) z[i] += gsig[i] * t[i];
}

// ---------------------------------------------------------------------------
// Host orchestration
// ---------------------------------------------------------------------------
extern "C" void kernel_launch(void* const* d_in, const int* in_sizes, int n_in,
                              void* d_out, int out_size, void* d_ws, size_t ws_size,
                              hipStream_t stream)
{
    (void)in_sizes; (void)n_in; (void)out_size; (void)ws_size;
    const int S = 128, N = 192, CM = 256, CZ = 128, HM = 8, HZ = 4;
    const int MA = S * N;        // 24576 rows of m
    const int MZ = N * N;        // 36864 rows of z
    const size_t A  = (size_t)MA * CM;   // 6,291,456 floats
    const size_t Zs = (size_t)MZ * CZ;   // 4,718,592 floats
    const float SCALE = 0.17677669529663687f; // 32^-0.5

    const float* in_m  = (const float*)d_in[0];
    const float* in_z  = (const float*)d_in[1];
    const float* zmask = (const float*)d_in[2];
    auto F = [&](int i) { return (const float*)d_in[i]; };

    float* out_m = (float*)d_out;
    float* out_z = out_m + A;
    float* ws = (float*)d_ws;

    hipMemcpyAsync(out_m, in_m, A * sizeof(float), hipMemcpyDeviceToDevice, stream);
    hipMemcpyAsync(out_z, in_z, Zs * sizeof(float), hipMemcpyDeviceToDevice, stream);

    auto ln = [&](const float* x, int bi, int gi, float* y, int rows, int C) {
        ln_kernel<<<dim3(rows), dim3(64), 0, stream>>>(x, F(gi), F(bi), y, C);
    };
    auto gemm = [&](const float* X, int wi, int bi, float* Y,
                    int M, int Nc, int K, int act, int accum, float alpha = 1.f) {
        gemm_kernel<<<dim3((M + 63) / 64, (Nc + 31) / 32, 1), 128, 0, stream>>>(
            X, F(wi), bi >= 0 ? F(bi) : nullptr, Y, M, Nc, K,
            (long)K, 1L, (long)Nc, 1L, (long)Nc, 1L,
            0L, 0L, 0L, 0L, 0L, 0L, 1, alpha, act, accum);
    };

    // ======================= msa_row (params 15..30) ========================
    {
        float* ml = ws;           float* zl = ml + A;
        float* q  = zl + Zs;      float* k  = q + A;
        float* v  = k + A;        float* g  = v + A;
        float* pb = g + A;        // MZ*HM
        ln(out_m, 19, 20, ml, MA, CM);
        ln(out_z, 21, 22, zl, MZ, CZ);
        gemm(ml, 28, 27, q, MA, CM, CM, 0, 0);
        gemm(ml, 18, 17, k, MA, CM, CM, 0, 0);
        gemm(ml, 30, 29, v, MA, CM, CM, 0, 0);
        gemm(ml, 16, 15, g, MA, CM, CM, 0, 0);
        gemm(zl, 26, 25, pb, MZ, HM, CZ, 0, 0);
        attn_kernel<<<dim3(S, N, HM), 64, 0, stream>>>(q, k, v, g, pb, zmask, q, N,
            (long)N * CM, (long)CM, (long)N * CM, 0L, (long)CM,
            0L, (long)N * HM, (long)HM, 0L, (long)N, 1L, SCALE);
        gemm(q, 24, 23, out_m, MA, CM, CM, 0, 1);
    }
    // ======================= msa_col (params 3..14) =========================
    {
        float* ml = ws; float* q = ml + A; float* k = q + A;
        float* v = k + A; float* g = v + A;
        ln(out_m, 7, 8, ml, MA, CM);
        gemm(ml, 12, 11, q, MA, CM, CM, 0, 0);
        gemm(ml,  6,  5, k, MA, CM, CM, 0, 0);
        gemm(ml, 14, 13, v, MA, CM, CM, 0, 0);
        gemm(ml,  4,  3, g, MA, CM, CM, 0, 0);
        attn_kernel<<<dim3(S, N, HM), 64, 0, stream>>>(q, k, v, g, nullptr, nullptr, q, S,
            (long)N * CM, (long)CM, 0L, (long)CM, (long)N * CM,
            0L, 0L, 0L, 0L, 0L, 0L, SCALE);
        gemm(q, 10, 9, out_m, MA, CM, CM, 0, 1);
    }
    // ======================= msa transition (31..36) ========================
    {
        float* ml = ws; float* h1 = ml + A; // MA*1024
        ln(out_m, 31, 32, ml, MA, CM);
        gemm(ml, 34, 33, h1, MA, 1024, CM, 1, 0);
        gemm(h1, 36, 35, out_m, MA, CM, 1024, 0, 1);
    }
    // ======================= opm (37..44), chunked over i ===================
    {
        float* ml = ws;
        float* bufA = ml + A;
        float* bufB = bufA + (size_t)MA * 32;
        float* outer = bufB + (size_t)MA * 32; // ICH*N*1024 floats
        ln(out_m, 41, 42, ml, MA, CM);
        gemm(ml, 38, 37, bufA, MA, 32, CM, 0, 0);
        gemm(ml, 40, 39, bufB, MA, 32, CM, 0, 0);
        const int ICH = 48;
        for (int i0 = 0; i0 < N; i0 += ICH) {
            // outer[i,j][c,d] = (1/S) sum_s a[s,i,c] * b[s,j,d]
            gemm_kernel<<<dim3(1, 1, ICH * N), 128, 0, stream>>>(
                bufA + (size_t)i0 * 32, bufB, nullptr, outer,
                32, 32, S,
                1L, (long)N * 32,          // X: c stride, s stride
                (long)N * 32, 1L,          // W: s stride, d stride
                32L, 1L,                   // Y: c stride, d stride
                0L, 32L,                   // xB0(j)=0, xB1(i)=32
                32L, 0L,                   // wB0(j)=32, wB1(i)=0
                1024L, (long)N * 1024,     // yB0(j), yB1(i)
                N, 1.f / S, 0, 0);
            gemm(outer, 44, 43, out_z + (size_t)i0 * N * CZ, ICH * N, CZ, 1024, 0, 1);
        }
    }
    // ================ triangle multiplicative updates =======================
    auto tri_update = [&](int base, bool incoming) {
        // base+0 ag.b, +1 ag.w, +2 ap.b, +3 ap.w, +4 bg.b, +5 bg.w, +6 bp.b,
        // +7 bp.w, +8 g.b, +9 g.w, +10 ln.b, +11 ln.g, +12 lnx.b, +13 lnx.g,
        // +14 o.b, +15 o.w
        float* zl = ws;       float* pa = zl + Zs;  float* pbuf = pa + Zs;
        float* gg = pbuf + Zs; float* xx = gg + Zs;  float* xl = xx + Zs;
        long tot = (long)MZ * CZ;
        int blks = (int)((tot + 255) / 256);
        ln(out_z, base + 10, base + 11, zl, MZ, CZ);
        gemm(zl, base + 3, base + 2, pa, MZ, CZ, CZ, 0, 0);
        gemm(zl, base + 1, base + 0, gg, MZ, CZ, CZ, 2, 0);
        gate_mask_mul<<<dim3(blks), 256, 0, stream>>>(pa, gg, zmask, CZ, tot);
        gemm(zl, base + 7, base + 6, pbuf, MZ, CZ, CZ, 0, 0);
        gemm(zl, base + 5, base + 4, gg, MZ, CZ, CZ, 2, 0);
        gate_mask_mul<<<dim3(blks), 256, 0, stream>>>(pbuf, gg, zmask, CZ, tot);
        long xRM = incoming ? (long)CZ : (long)N * CZ;
        long xRK = incoming ? (long)N * CZ : (long)CZ;
        long wRK = incoming ? (long)N * CZ : (long)CZ;
        long wRN = incoming ? (long)CZ : (long)N * CZ;
        // x[i,j,c] = sum_k a[.,.,c] * b[.,.,c]  (batched over channel c)
        gemm_kernel<<<dim3(3, 6, CZ), 128, 0, stream>>>(
            pa, pbuf, nullptr, xx, N, N, N,
            xRM, xRK, wRK, wRN, (long)N * CZ, (long)CZ,
            1L, 0L, 1L, 0L, 1L, 0L, CZ, 1.f, 0, 0);
        ln(xx, base + 12, base + 13, xl, MZ, CZ);
        gemm(xl, base + 15, base + 14, pa, MZ, CZ, CZ, 0, 0);
        gemm(zl, base + 9, base + 8, gg, MZ, CZ, CZ, 2, 0);
        gated_add<<<dim3(blks), 256, 0, stream>>>(out_z, gg, pa, tot);
    };
    tri_update(95, false);  // tri_out (outgoing)
    tri_update(79, true);   // tri_in  (incoming)

    // ==================== triangle attentions ===============================
    auto tri_attn = [&](int base, bool ending) {
        // base+0 g.b, +1 g.w, +2 k.b, +3 k.w, +4 ln.b, +5 ln.g, +6 o.b, +7 o.w,
        // +8 pb.b, +9 pb.w, +10 q.b, +11 q.w, +12 v.b, +13 v.w
        float* zl = ws;  float* q = zl + Zs; float* k = q + Zs;
        float* v = k + Zs; float* g = v + Zs; float* pbz = g + Zs;
        ln(out_z, base + 4, base + 5, zl, MZ, CZ);
        gemm(zl, base + 11, base + 10, q, MZ, CZ, CZ, 0, 0);
        gemm(zl, base + 3,  base + 2,  k, MZ, CZ, CZ, 0, 0);
        gemm(zl, base + 13, base + 12, v, MZ, CZ, CZ, 0, 0);
        gemm(zl, base + 1,  base + 0,  g, MZ, CZ, CZ, 0, 0);
        gemm(zl, base + 9,  base + 8,  pbz, MZ, HZ, CZ, 0, 0);
        if (!ending) {
            attn_kernel<<<dim3(N, N, HZ), 64, 0, stream>>>(q, k, v, g, pbz, zmask, q, N,
                (long)N * CZ, (long)CZ, (long)N * CZ, 0L, (long)CZ,
                0L, (long)N * HZ, (long)HZ, (long)N, 0L, 1L, SCALE);
        } else {
            attn_kernel<<<dim3(N, N, HZ), 64, 0, stream>>>(q, k, v, g, pbz, zmask, q, N,
                (long)CZ, (long)N * CZ, (long)CZ, 0L, (long)N * CZ,
                0L, (long)HZ, (long)N * HZ, 1L, 0L, (long)N, SCALE);
        }
        gemm(q, base + 7, base + 6, out_z, MZ, CZ, CZ, 0, 1);
    };
    tri_attn(65, false);  // ta_start
    tri_attn(51, true);   // ta_end

    // ==================== pair transition (45..50) ==========================
    {
        float* zl = ws; float* h1 = zl + Zs; // MZ*512
        ln(out_z, 45, 46, zl, MZ, CZ);
        gemm(zl, 48, 47, h1, MZ, 512, CZ, 1, 0);
        gemm(h1, 50, 49, out_z, MZ, CZ, 512, 0, 1);
    }
}
